// BiLSTM_CRF_56298431316128
// MI455X (gfx1250) — matrix operand order repaired and checked by reading, not compiled, not run on hardware
//
#include <hip/hip_runtime.h>
#include <hip/hip_bf16.h>

// ---------------------------------------------------------------------------
// BiLSTM-CRF on gfx1250 (MI455X), wave32 + WMMA bf16 (f32 accum) + TDM.
#define Bn   32
#define Tn   256
#define En   256
#define Hn   256          // per-direction hidden
#define G4H  1024         // 4*H
#define Kt   12           // tagset
#define START_TAG 9
#define STOP_TAG  10
#define NEGV (-10000.0f)

#define LDP  264          // padded LDS row stride (bf16 elems): 132 dw % 64 banks = 4

typedef __bf16 bf16;
typedef __attribute__((ext_vector_type(16))) bf16  bf16x16;
typedef __attribute__((ext_vector_type(8)))  float f32x8;
typedef __attribute__((ext_vector_type(4)))  unsigned v4u;
typedef __attribute__((ext_vector_type(4)))  int      v4i;
typedef __attribute__((ext_vector_type(8)))  int      v8i;

#if defined(__HIP_DEVICE_COMPILE__) && __has_builtin(__builtin_amdgcn_tensor_load_to_lds) && __has_builtin(__builtin_amdgcn_s_wait_tensorcnt)
#define HAVE_TDM 1
#else
#define HAVE_TDM 0
#endif

// ---- workspace layout (bytes) ---------------------------------------------
// WihBf [2][1024][256] bf16
// WhhBf [2][1024][256] bf16
// biasG [2][1024] f32
// xg    [T*B][256] bf16           (row = t*32+b)
// Gin   [2][T][1024][B] f32       (TRANSPOSED: fragment-friendly)
// hbuf  [T][512][B] f32           (TRANSPOSED concat fwd|bwd hidden)
// feats [T*B][12] f32
// bp    [T*B][12] i32
static const size_t OFF_WIHBF = 0;
static const size_t OFF_WHHBF = OFF_WIHBF + (size_t)2*G4H*Hn*2;
static const size_t OFF_BIAS  = OFF_WHHBF + (size_t)2*G4H*Hn*2;
static const size_t OFF_XG    = OFF_BIAS  + (size_t)2*G4H*4;
static const size_t OFF_GIN   = OFF_XG    + (size_t)Tn*Bn*En*2;
static const size_t OFF_HBUF  = OFF_GIN   + (size_t)2*Tn*Bn*G4H*4;
static const size_t OFF_FEATS = OFF_HBUF  + (size_t)Tn*Bn*512*4;
static const size_t OFF_BP    = OFF_FEATS + (size_t)Tn*Bn*Kt*4;

// LDS carve inside k_recurrent (dynamic shared):
//   whsh: 512 rows x 264 bf16 (persistent Whh gates i,f) = 270336 B
//   hsh : 2 x 32 rows x 264 bf16 (h double buffer)        =  33792 B
#define WHSH_ELEMS (512 * LDP)
#define HSH_STRIDE (Bn * LDP)
#define SMEM_BYTES ((WHSH_ELEMS + 2 * HSH_STRIDE) * 2)

// ---- small helpers --------------------------------------------------------
__device__ inline bf16 f2bf(float f) {
  unsigned u = __float_as_uint(f);
  unsigned short h = (unsigned short)((u + 0x7FFFu + ((u >> 16) & 1u)) >> 16);
  return __builtin_bit_cast(bf16, h);
}
__device__ inline float sigm(float x) { return 1.0f / (1.0f + __expf(-x)); }
__device__ inline float tanh_fast(float x) { return 1.0f - 2.0f / (__expf(2.0f * x) + 1.0f); }

// 16-bit A-matrix 16x32 fragment (ISA 7.12.2): lane<16 -> K=[k0..k0+7],[k0+16..k0+23]
// lane>=16 -> K=[k0+8..k0+15],[k0+24..k0+31]; M = m0 + (lane&15).
template <typename P>
__device__ inline bf16x16 load_a_frag(P base, int ld, int m0, int k0, int lane) {
  int l15 = lane & 15, hh = (lane >> 4) << 3;
  P p0 = base + (m0 + l15) * ld + k0 + hh;
  P p1 = p0 + 16;
  bf16x16 a;
#pragma unroll
  for (int i = 0; i < 8; ++i) { a[i] = p0[i]; a[8 + i] = p1[i]; }
  return a;
}

// 16-bit B-matrix 32x16 fragment: lane<16 holds K=[k0..k0+7],[k0+16..k0+23] of column n0+lane,
// lane>=16 holds K=[k0+8..k0+15],[k0+24..k0+31] of column n0+lane-16. B[k,n] = W[n][k] row-major.
template <typename P>
__device__ inline bf16x16 load_b_frag(P W, int ld, int n0, int k0, int lane) {
  int l15 = lane & 15, hh = (lane >> 4) << 3;
  P p = W + (n0 + l15) * ld + k0 + hh;
  bf16x16 b;
#pragma unroll
  for (int i = 0; i < 8; ++i) { b[i] = p[i]; b[8 + i] = p[16 + i]; }
  return b;
}

// ---- prep kernels ---------------------------------------------------------
__global__ void k_convert(const float* __restrict__ src, bf16* __restrict__ dst, int n) {
  for (int i = blockIdx.x * blockDim.x + threadIdx.x; i < n; i += gridDim.x * blockDim.x)
    dst[i] = f2bf(src[i]);
}

__global__ void k_bias(const float* bihf, const float* bhhf,
                       const float* bihb, const float* bhhb, float* biasG) {
  int tid = blockIdx.x * blockDim.x + threadIdx.x;
  if (tid < 2 * G4H) {
    int d = tid >> 10, j = tid & 1023;
    biasG[tid] = d ? (bihb[j] + bhhb[j]) : (bihf[j] + bhhf[j]);
  }
}

__global__ void k_gather(const int* __restrict__ sent, const float* __restrict__ emb,
                         bf16* __restrict__ xg) {
  int row = blockIdx.x;            // t*32 + b
  int t = row >> 5, b = row & 31;
  int tok = sent[b * Tn + t];
  int e = threadIdx.x;
  xg[(size_t)row * En + e] = f2bf(emb[(size_t)tok * En + e]);
}

// ---- input projection: Gin[d][t][n][b] = (xg @ Wih_d^T + bias) transposed --
__global__ __launch_bounds__(256) void k_gemm_in(const bf16* __restrict__ xg,
                                                 const bf16* __restrict__ WihBf,
                                                 const float* __restrict__ biasG,
                                                 float* __restrict__ Gin) {
  int bx = blockIdx.x;             // 2 dirs * 512 mtiles * 8 ngroups = 8192 blocks
  int dir = bx >> 12;
  int mt  = (bx & 4095) >> 3;
  int ng  = bx & 7;
  int lane = threadIdx.x & 31, wv = threadIdx.x >> 5;
  int m0 = mt * 16;
  int n0 = (ng * 8 + wv) * 16;
  int l15 = lane & 15, h8 = (lane >> 4) * 8;

  const bf16* W = WihBf + (size_t)dir * G4H * En;
  float bv = biasG[dir * G4H + n0 + l15];
  f32x8 acc;
#pragma unroll
  for (int v = 0; v < 8; ++v) acc[v] = bv;

#pragma unroll
  for (int kc = 0; kc < 8; ++kc) {
    bf16x16 a = load_a_frag(xg, En, m0, kc * 32, lane);
    bf16x16 b = load_b_frag(W, En, n0, kc * 32, lane);
    acc = __builtin_amdgcn_wmma_f32_16x16x32_bf16(false, a, false, b,
                                                  (short)0, acc, false, false);
  }
  // an M-tile spans a single timestep: rows = t*32 + b
  int t  = m0 >> 5;
  int b0 = (m0 & 31) + h8;         // batch base for this lane's 8 values
  float* out = Gin + (((size_t)dir * Tn + t) * G4H + n0 + l15) * Bn + b0;
#pragma unroll
  for (int v = 0; v < 8; ++v) out[v] = acc[v];   // contiguous -> wide stores
}

// ---- recurrent scan: one persistent workgroup per direction ---------------
__global__ __launch_bounds__(512) void k_recurrent(const float* __restrict__ Gin,
                                                   const bf16* __restrict__ WhhBf,
                                                   const float* __restrict__ h0,
                                                   const float* __restrict__ c0,
                                                   float* __restrict__ hbuf) {
  int dir = blockIdx.x;
  int tid = threadIdx.x, lane = tid & 31, wv = tid >> 5;  // 16 waves
  int l15 = lane & 15, h8 = (lane >> 4) * 8;
  int col = wv * 16 + l15;                                 // hidden index 0..255

  extern __shared__ char smem[];
  bf16* whsh = (bf16*)smem;                          // Whh rows 0..511 (i,f gates)
  bf16* hsh  = (bf16*)(smem + (size_t)WHSH_ELEMS * 2);

  const bf16* Whh = WhhBf + (size_t)dir * G4H * Hn;

  // --- stage Whh gate-i/f half into LDS (padded rows) ---
#if HAVE_TDM
  if (wv == 0) {
    unsigned lds_base = (unsigned)(size_t)whsh;      // low 32 bits = LDS offset
    unsigned long long ga = (unsigned long long)(size_t)Whh;
    v4u g0;
    g0[0] = 1u;                                       // count=1 (valid descriptor)
    g0[1] = lds_base;                                 // lds_addr
    g0[2] = (unsigned)ga;                             // global_addr[31:0]
    g0[3] = (unsigned)((ga >> 32) & 0x01FFFFFFull) | 0x80000000u;  // addr[56:32] | type=2
    v8i g1;
    g1[0] = 0x07910000;            // data_size=2B, pad_enable, interval=128dw, amount=4dw
    g1[1] = (int)(256u << 16);     // tensor_dim0 = 256 (low16)
    g1[2] = (int)(512u << 16);     // tensor_dim1 = 512 (low16)
    g1[3] = (int)(256u << 16);     // tile_dim0 = 256
    g1[4] = 512;                   // tile_dim1 = 512, tile_dim2 = 0
    g1[5] = 256;                   // tensor_dim0_stride = 256
    g1[6] = 0; g1[7] = 0;
    v4i gz; gz[0] = gz[1] = gz[2] = gz[3] = 0;
#if __clang_major__ >= 23
    v8i z8; z8[0]=z8[1]=z8[2]=z8[3]=z8[4]=z8[5]=z8[6]=z8[7]=0;
    __builtin_amdgcn_tensor_load_to_lds(g0, g1, gz, gz, z8, 0);
#else
    __builtin_amdgcn_tensor_load_to_lds(g0, g1, gz, gz, 0);
#endif
    __builtin_amdgcn_s_wait_tensorcnt(0);
  }
#else
  for (int i = tid; i < 512 * Hn; i += 512) {
    int r = i >> 8, c = i & 255;
    whsh[r * LDP + c] = Whh[i];
  }
#endif

  // init h (bf16, padded row-major [B][LDP]) and c-state in C-fragment layout
  for (int i = tid; i < Bn * Hn; i += 512) {
    int r = i >> 8, c = i & 255;
    hsh[r * LDP + c] = f2bf(h0[(size_t)dir * Bn * Hn + i]);
  }
  f32x8 cst[2];
#pragma unroll
  for (int mi = 0; mi < 2; ++mi)
#pragma unroll
    for (int v = 0; v < 8; ++v) {
      int row = mi * 16 + h8 + v;
      cst[mi][v] = c0[(size_t)dir * Bn * Hn + (size_t)row * Hn + col];
    }
  __syncthreads();

  for (int s = 0; s < Tn; ++s) {
    int t = dir ? (Tn - 1 - s) : s;
    bf16* hcur = hsh + (s & 1) * HSH_STRIDE;
    bf16* hnxt = hsh + ((s & 1) ^ 1) * HSH_STRIDE;

    // A fragments from LDS h (shared by all tiles of this wave)
    bf16x16 areg[2][8];
#pragma unroll
    for (int mi = 0; mi < 2; ++mi)
#pragma unroll
      for (int kc = 0; kc < 8; ++kc)
        areg[mi][kc] = load_a_frag(hcur, LDP, mi * 16, kc * 32, lane);

    // accumulators init from transposed Gin: 8 contiguous floats per (mi,ga)
    const float* gbase = Gin + (((size_t)dir * Tn + t) * G4H) * Bn;
    f32x8 acc[2][4];
#pragma unroll
    for (int mi = 0; mi < 2; ++mi)
#pragma unroll
      for (int ga = 0; ga < 4; ++ga) {
        const float* p = gbase + (size_t)(ga * Hn + col) * Bn + mi * 16 + h8;
#pragma unroll
        for (int v = 0; v < 8; ++v) acc[mi][ga][v] = p[v];
      }

    if (s < Tn - 1) {  // prefetch next step's Gin (global_prefetch_b8)
      int t2 = dir ? (Tn - 2 - s) : (s + 1);
      __builtin_prefetch(Gin + (((size_t)dir * Tn + t2) * G4H + col) * Bn, 0, 1);
    }

    // g += h @ Whh^T : gates i,f from LDS-resident half; g,o streamed from L2
#pragma unroll
    for (int ga = 0; ga < 4; ++ga) {
      bf16x16 breg[8];
      if (ga < 2) {
#pragma unroll
        for (int kc = 0; kc < 8; ++kc)
          breg[kc] = load_b_frag(whsh, LDP, ga * Hn + wv * 16, kc * 32, lane);
      } else {
#pragma unroll
        for (int kc = 0; kc < 8; ++kc)
          breg[kc] = load_b_frag(Whh, Hn, ga * Hn + wv * 16, kc * 32, lane);
      }
#pragma unroll
      for (int mi = 0; mi < 2; ++mi)
#pragma unroll
        for (int kc = 0; kc < 8; ++kc)
          acc[mi][ga] = __builtin_amdgcn_wmma_f32_16x16x32_bf16(
              false, areg[mi][kc], false, breg[kc], (short)0, acc[mi][ga],
              false, false);
    }

    // elementwise cell update (gate order i,f,g,o); h -> LDS (bf16) + global (f32)
    int ch = dir * Hn + col;
#pragma unroll
    for (int mi = 0; mi < 2; ++mi) {
      float hv[8];
#pragma unroll
      for (int v = 0; v < 8; ++v) {
        float gi = acc[mi][0][v], gf = acc[mi][1][v];
        float gg = acc[mi][2][v], go = acc[mi][3][v];
        float c = sigm(gf) * cst[mi][v] + sigm(gi) * tanh_fast(gg);
        hv[v] = sigm(go) * tanh_fast(c);
        cst[mi][v] = c;
      }
      float* hb = hbuf + ((size_t)t * 512 + ch) * Bn + mi * 16 + h8;
#pragma unroll
      for (int v = 0; v < 8; ++v) {
        hb[v] = hv[v];                                     // contiguous in batch
        hnxt[(mi * 16 + h8 + v) * LDP + col] = f2bf(hv[v]);
      }
    }
    __syncthreads();
  }
}

// ---- hidden2tag (hbuf transposed: [t][ch][b]) ------------------------------
__global__ void k_feats(const float* __restrict__ hbuf, const float* __restrict__ Wout,
                        const float* __restrict__ bout, float* __restrict__ feats) {
  int tid = blockIdx.x * blockDim.x + threadIdx.x;   // 8192 rows * 12 tags
  if (tid < Tn * Bn * Kt) {
    int r = tid / Kt, k = tid % Kt;
    int t = r >> 5, b = r & 31;
    const float* hr = hbuf + (size_t)t * 512 * Bn + b;
    const float* wr = Wout + (size_t)k * 512;
    float s = bout[k];
#pragma unroll 8
    for (int j = 0; j < 512; ++j) s += hr[(size_t)j * Bn] * wr[j];
    feats[tid] = s;
  }
}

// ---- viterbi decode (K=12, B=32): one block, 384 threads -------------------
__global__ void k_viterbi(const float* __restrict__ feats, const float* __restrict__ trans,
                          int* __restrict__ bp, float* __restrict__ out) {
  int tid = threadIdx.x;                 // 384 = 32 b * 12 k
  int b = tid / Kt, k = tid % Kt;
  __shared__ float fv[Bn * Kt];
  __shared__ float term[Bn * Kt];
  __shared__ float bsc[Bn];
  __shared__ int   barg[Bn];

  fv[tid] = (k == START_TAG) ? 0.0f : NEGV;
  __syncthreads();

  for (int t = 0; t < Tn; ++t) {
    float best = -3.0e38f; int arg = 0;
    const float* fvb = &fv[b * Kt];
    const float* trk = &trans[k * Kt];
#pragma unroll
    for (int p = 0; p < Kt; ++p) {
      float sc = fvb[p] + trk[p];
      if (sc > best) { best = sc; arg = p; }   // first-max (jnp.argmax semantics)
    }
    int idx = (t * Bn + b) * Kt + k;
    bp[idx] = arg;
    float nv = best + feats[idx];
    __syncthreads();
    fv[b * Kt + k] = nv;
    __syncthreads();
  }

  term[tid] = fv[b * Kt + k] + trans[STOP_TAG * Kt + k];
  __syncthreads();
  if (k == 0) {
    float best = -3.0e38f; int arg = 0;
#pragma unroll
    for (int p = 0; p < Kt; ++p) {
      float sc = term[b * Kt + p];
      if (sc > best) { best = sc; arg = p; }
    }
    bsc[b] = best; barg[b] = arg;
  }
  __syncthreads();

  if (tid < Bn) {
    int bb = tid;
    out[bb] = bsc[bb];                       // path scores [B]
    int tag = barg[bb];
    for (int t = Tn - 1; t >= 0; --t) {      // best path [B,T] (as float)
      out[Bn + bb * Tn + t] = (float)tag;
      tag = bp[(t * Bn + bb) * Kt + tag];
    }
  }
}

// ---------------------------------------------------------------------------
extern "C" void kernel_launch(void* const* d_in, const int* in_sizes, int n_in,
                              void* d_out, int out_size, void* d_ws, size_t ws_size,
                              hipStream_t stream) {
  (void)in_sizes; (void)n_in; (void)out_size; (void)ws_size;
  const int*   sent  = (const int*)  d_in[0];
  const float* emb   = (const float*)d_in[1];
  const float* Wihf  = (const float*)d_in[2];
  const float* Whhf  = (const float*)d_in[3];
  const float* bihf  = (const float*)d_in[4];
  const float* bhhf  = (const float*)d_in[5];
  const float* Wihb  = (const float*)d_in[6];
  const float* Whhb  = (const float*)d_in[7];
  const float* bihb  = (const float*)d_in[8];
  const float* bhhb  = (const float*)d_in[9];
  const float* Wout  = (const float*)d_in[10];
  const float* bout  = (const float*)d_in[11];
  const float* h0    = (const float*)d_in[12];
  const float* c0    = (const float*)d_in[13];
  const float* trans = (const float*)d_in[14];
  float* out = (float*)d_out;

  char* ws = (char*)d_ws;
  bf16*  WihBf = (bf16*)(ws + OFF_WIHBF);
  bf16*  WhhBf = (bf16*)(ws + OFF_WHHBF);
  float* biasG = (float*)(ws + OFF_BIAS);
  bf16*  xg    = (bf16*)(ws + OFF_XG);
  float* Gin   = (float*)(ws + OFF_GIN);
  float* hbuf  = (float*)(ws + OFF_HBUF);
  float* feats = (float*)(ws + OFF_FEATS);
  int*   bp    = (int*)(ws + OFF_BP);

  const int NW = G4H * Hn;  // 262144 elements per weight matrix
  k_convert<<<512, 256, 0, stream>>>(Wihf, WihBf,       NW);
  k_convert<<<512, 256, 0, stream>>>(Wihb, WihBf + NW,  NW);
  k_convert<<<512, 256, 0, stream>>>(Whhf, WhhBf,       NW);
  k_convert<<<512, 256, 0, stream>>>(Whhb, WhhBf + NW,  NW);
  k_bias   <<<8, 256, 0, stream>>>(bihf, bhhf, bihb, bhhb, biasG);
  k_gather <<<Tn * Bn, En, 0, stream>>>(sent, emb, xg);
  k_gemm_in<<<8192, 256, 0, stream>>>(xg, WihBf, biasG, Gin);
  k_recurrent<<<2, 512, SMEM_BYTES, stream>>>(Gin, WhhBf, h0, c0, hbuf);
  k_feats  <<<384, 256, 0, stream>>>(hbuf, Wout, bout, feats);
  k_viterbi<<<1, 384, 0, stream>>>(feats, trans, bp, out);
}
